// LoRAAttention_54047868453051
// MI455X (gfx1250) — matrix operand order repaired and checked by reading, not compiled, not run on hardware
//
#include <hip/hip_runtime.h>
#include <hip/hip_bf16.h>
#include <cstdint>

typedef __bf16 bf16_t;
typedef __attribute__((ext_vector_type(16))) __bf16 v16bf;
typedef __attribute__((ext_vector_type(8)))  float  v8f;
typedef __attribute__((ext_vector_type(4)))  unsigned int u32x4;
typedef __attribute__((ext_vector_type(8)))  int i32x8;
typedef __attribute__((ext_vector_type(4)))  int i32x4;

#define WMMA_BF16(a, b, c) \
  __builtin_amdgcn_wmma_f32_16x16x32_bf16(false, (a), false, (b), (short)0, (c), false, false)

constexpr int Bb   = 8;
constexpr int Nn   = 1024;
constexpr int Cc   = 768;
constexpr int Hh   = 12;
constexpr int HD   = 64;
constexpr int RANK = 64;
constexpr int MN   = Bb * Nn;            // 8192 rows
constexpr float ATTN_SCALE = 0.125f;     // 64^-0.5
constexpr float LORA_SCALE = 1.0f / 64.0f;

// ---------------------------------------------------------------- helpers ---

__global__ void cvt_f32_bf16(const float* __restrict__ in, bf16_t* __restrict__ out, int n) {
  int i = blockIdx.x * blockDim.x + threadIdx.x;
  if (i < n) out[i] = (bf16_t)in[i];
}

// TDM: DMA a 2D tile [h rows x w elems] of bf16 (row stride ld elems) into LDS.
// D# per CDNA5 ISA: group0 = count|lds_addr|global_addr|type=2,
// group1 = data_size + tensor/tile dims + dim0 stride. Groups 2/3 zero (2-D tile).
__device__ __forceinline__ void tdm_load_2d(bf16_t* lds_dst, const bf16_t* gsrc,
                                            int ld, int w, int h) {
  union { unsigned long long q[2]; u32x4 v; } g0;
  union { unsigned long long q[4]; i32x8 v; } g1;
  unsigned long long ga = (unsigned long long)(uintptr_t)gsrc;          // byte addr
  unsigned long long la = (unsigned)(uintptr_t)lds_dst;                 // LDS = addr[31:0]
  g0.q[0] = 1ull | (la << 32);                                          // count=1, lds_addr
  g0.q[1] = (ga & 0x01FFFFFFFFFFFFFFull) | (2ull << 62);                // global_addr, type=2
  unsigned long long W = (unsigned long long)(unsigned)w;
  unsigned long long H = (unsigned long long)(unsigned)h;
  unsigned long long S = (unsigned long long)(unsigned)ld;              // data_size units
  g1.q[0] = (1ull << 16) | ((W & 0xFFFFull) << 48);        // mask=0, data_size=1(2B), td0 lo
  g1.q[1] = (W >> 16) | (H << 16) | ((W & 0xFFFFull) << 48); // td0 hi, td1, tile_dim0
  g1.q[2] = H | ((S & 0xFFFFFFFFull) << 32);               // tile_dim1, tile_dim2=0, stride lo
  g1.q[3] = (S >> 32);                                     // stride hi, dim1_stride=0 (2D)
  i32x4 z4 = {0, 0, 0, 0};
  i32x8 z8 = {0, 0, 0, 0, 0, 0, 0, 0};
  __builtin_amdgcn_tensor_load_to_lds(g0.v, g1.v, z4, z4, z8, 0);
}

// A fragment: 16x32 bf16 (MxK), row-major global, leading dim ld.
// lanes 0-15/16-31 hold rows M=0..15; e<8: K=e+8*hf ; e>=8: K=e+8+8*hf
__device__ __forceinline__ v16bf load_a_frag(const bf16_t* __restrict__ base, int ld,
                                             int mb, int kb) {
  int l  = threadIdx.x & 31;
  int hf = l >> 4;
  const bf16_t* row = base + (size_t)(mb + (l & 15)) * ld + kb;
  union { uint4 u[2]; v16bf v; } t;
  t.u[0] = *reinterpret_cast<const uint4*>(row + 8 * hf);
  t.u[1] = *reinterpret_cast<const uint4*>(row + 16 + 8 * hf);
  return t.v;
}

// B fragment from an LDS tile laid out [n][ldl] (k contiguous per row):
// B(k,n): e<8: K=e+8*hf ; e>=8: K=e+8+8*hf; n = lane&15.
__device__ __forceinline__ v16bf load_b_frag_lds(const bf16_t* tile, int ldl,
                                                 int nb, int kb) {
  int l  = threadIdx.x & 31;
  int hf = l >> 4;
  const bf16_t* p = tile + (size_t)(nb + (l & 15)) * ldl + kb + 8 * hf;
  union { uint4 u[2]; v16bf v; } t;
  t.u[0] = *reinterpret_cast<const uint4*>(p);
  t.u[1] = *reinterpret_cast<const uint4*>(p + 16);
  return t.v;
}

// ------------------------------------------------------------- GEMM (WMMA) ---
// C[M,N] = scale * (A[M,K] @ W[N,K]^T) (+ bias[n]) (+= existing Cf if accumulate)
// Block = 8 waves = 128(M) x 64(N) tile. W tile (64x32) is TDM-staged into LDS
// once per block per K-step and shared by all waves.
__global__ void gemm_bf16_wmma(const bf16_t* __restrict__ A, int lda,
                               const bf16_t* __restrict__ W, int ldw,
                               float* __restrict__ Cf, bf16_t* __restrict__ Cb, int ldc,
                               const float* __restrict__ bias, float scale,
                               int K, int accumulate) {
  __shared__ bf16_t sW[64 * 32];           // [n_local][32]
  int wave = threadIdx.x >> 5;
  int lane = threadIdx.x & 31;
  int mb = blockIdx.x * 128 + wave * 16;
  int nb = blockIdx.y * 64;

  v8f acc[4] = {{0,0,0,0,0,0,0,0},{0,0,0,0,0,0,0,0},
                {0,0,0,0,0,0,0,0},{0,0,0,0,0,0,0,0}};

  for (int k = 0; k < K; k += 32) {
    __syncthreads();                       // previous tile fully consumed
    if (wave == 0) {
      tdm_load_2d(sW, W + (size_t)nb * ldw + k, ldw, 32, 64);
      __builtin_amdgcn_s_wait_tensorcnt(0);
    }
    __syncthreads();                       // tile visible to all waves
    v16bf a = load_a_frag(A, lda, mb, k);
#pragma unroll
    for (int t = 0; t < 4; ++t) {
      v16bf w = load_b_frag_lds(sW, 32, t * 16, 0);
      acc[t] = WMMA_BF16(a, w, acc[t]);
    }
  }

  int hf = lane >> 4, nc = lane & 15;
#pragma unroll
  for (int t = 0; t < 4; ++t) {
#pragma unroll
    for (int r = 0; r < 8; ++r) {
      int row = mb + r + 8 * hf;
      int col = nb + t * 16 + nc;
      float v = acc[t][r] * scale;
      if (bias) v += bias[col];
      size_t idx = (size_t)row * ldc + col;
      if (accumulate) v += Cf[idx];
      if (Cb) Cb[idx] = (bf16_t)v;
      else    Cf[idx] = v;
    }
  }
}

// ------------------------------------------------------------------- pack ---
__global__ void pack_qkv(const float* __restrict__ qkv,
                         bf16_t* __restrict__ qh, bf16_t* __restrict__ kh,
                         bf16_t* __restrict__ vh) {
  int i = blockIdx.x * blockDim.x + threadIdx.x;
  if (i >= MN * Cc) return;
  int c = i % Cc;
  int n = (i / Cc) % Nn;
  int b = i / (Cc * Nn);
  int h = c / HD, d = c % HD;
  size_t src = (size_t)(b * Nn + n) * (3 * Cc);
  size_t dst = ((size_t)(b * Hh + h) * Nn + n) * HD + d;
  qh[dst] = (bf16_t)(qkv[src + c] * ATTN_SCALE);
  kh[dst] = (bf16_t)(qkv[src + Cc + c]);
  vh[dst] = (bf16_t)(qkv[src + 2 * Cc + c]);
}

// -------------------------------------------------------- flash attention ---
// grid: (N/64, B*H); block: 128 threads (4 waves). All 4 waves share one (b,h)
// and iterate the same 32-key tiles: K tile TDM-staged to LDS row-major,
// V tile staged transposed so P@V B-fragments are contiguous ds_load_b128.
__global__ void flash_attn(const bf16_t* __restrict__ Q, const bf16_t* __restrict__ Kh,
                           const bf16_t* __restrict__ Vh, bf16_t* __restrict__ O) {
  __shared__ bf16_t sK[32][64];            // [key][dim]
  __shared__ bf16_t sVT[64][32];           // [dim][key]  (transposed)
  __shared__ bf16_t ptile[4][16][32];      // per-wave P round-trip
  int wave = threadIdx.x >> 5;
  int lane = threadIdx.x & 31;
  int hf   = lane >> 4;
  int lan  = lane & 15;
  int bh = blockIdx.y;
  int b = bh / Hh, h = bh % Hh;
  int qrow = blockIdx.x * 64 + wave * 16;

  const bf16_t* Qp = Q  + (size_t)bh * Nn * HD;
  const bf16_t* Kp = Kh + (size_t)bh * Nn * HD;
  const bf16_t* Vp = Vh + (size_t)bh * Nn * HD;

  v16bf aq0 = load_a_frag(Qp, HD, qrow, 0);
  v16bf aq1 = load_a_frag(Qp, HD, qrow, 32);

  v8f acc[4] = {{0,0,0,0,0,0,0,0},{0,0,0,0,0,0,0,0},
                {0,0,0,0,0,0,0,0},{0,0,0,0,0,0,0,0}};
  float m_i[8], l_i[8];
#pragma unroll
  for (int r = 0; r < 8; ++r) { m_i[r] = -3.0e38f; l_i[r] = 0.0f; }

  for (int j = 0; j < Nn; j += 32) {
    __syncthreads();                       // previous K/V tiles consumed
    if (wave == 0)                          // K tile 32 rows x 64 dims via TDM
      tdm_load_2d(&sK[0][0], Kp + (size_t)j * HD, HD, 64, 32);
    // V tile staged transposed by all 128 threads (2 x uint4 each)
#pragma unroll
    for (int pass = 0; pass < 2; ++pass) {
      int idx = (threadIdx.x + pass * 128) * 8;      // 2048 elems total
      int key = idx / HD, d0 = idx % HD;
      union { uint4 u; bf16_t e[8]; } tv_;
      tv_.u = *reinterpret_cast<const uint4*>(Vp + (size_t)(j + key) * HD + d0);
#pragma unroll
      for (int e = 0; e < 8; ++e) sVT[d0 + e][key] = tv_.e[e];
    }
    if (wave == 0) __builtin_amdgcn_s_wait_tensorcnt(0);
    __syncthreads();                       // tiles ready

    // S = Q @ K^T : two 16x16 column tiles, K-dim 64 = 2 wmma each
    v8f s[2] = {{0,0,0,0,0,0,0,0},{0,0,0,0,0,0,0,0}};
#pragma unroll
    for (int t = 0; t < 2; ++t) {
      v16bf k0 = load_b_frag_lds(&sK[0][0], HD, t * 16, 0);
      s[t] = WMMA_BF16(aq0, k0, s[t]);
      v16bf k1 = load_b_frag_lds(&sK[0][0], HD, t * 16, 32);
      s[t] = WMMA_BF16(aq1, k1, s[t]);
    }
    // online softmax; row r+8*hf lives in a 16-lane group -> width-16 butterfly
#pragma unroll
    for (int r = 0; r < 8; ++r) {
      float tm = fmaxf(s[0][r], s[1][r]);
#pragma unroll
      for (int m = 8; m >= 1; m >>= 1) tm = fmaxf(tm, __shfl_xor(tm, m, 16));
      float nm   = fmaxf(m_i[r], tm);
      float corr = __expf(m_i[r] - nm);
      m_i[r] = nm;
      float p0 = __expf(s[0][r] - nm);
      float p1 = __expf(s[1][r] - nm);
      s[0][r] = p0; s[1][r] = p1;
      float ts = p0 + p1;
#pragma unroll
      for (int m = 8; m >= 1; m >>= 1) ts += __shfl_xor(ts, m, 16);
      l_i[r] = l_i[r] * corr + ts;
#pragma unroll
      for (int t = 0; t < 4; ++t) acc[t][r] *= corr;
    }
    // C-layout P -> LDS -> A-layout bf16 fragment (wave-private tile)
#pragma unroll
    for (int t = 0; t < 2; ++t)
#pragma unroll
      for (int r = 0; r < 8; ++r)
        ptile[wave][r + 8 * hf][t * 16 + lan] = (bf16_t)s[t][r];
    asm volatile("s_wait_dscnt 0" ::: "memory");  // wave-local LDS RAW fence
    v16bf ap;
    {
      const bf16_t* prow = &ptile[wave][lan][0];
#pragma unroll
      for (int e = 0; e < 8; ++e)  ap[e] = prow[e + 8 * hf];
#pragma unroll
      for (int e = 8; e < 16; ++e) ap[e] = prow[e + 8 + 8 * hf];
    }
    // O += P @ V : V B-fragments contiguous from transposed LDS tile
#pragma unroll
    for (int t = 0; t < 4; ++t) {
      v16bf bv = load_b_frag_lds(&sVT[0][0], 32, t * 16, 0);
      acc[t] = WMMA_BF16(ap, bv, acc[t]);
    }
  }
  // normalize + store into [B, N, C] (heads concatenated along C)
#pragma unroll
  for (int t = 0; t < 4; ++t)
#pragma unroll
    for (int r = 0; r < 8; ++r) {
      int row = qrow + r + 8 * hf;
      float v = acc[t][r] / l_i[r];
      O[((size_t)(b * Nn) + row) * Cc + h * HD + t * 16 + lan] = (bf16_t)v;
    }
}

// ----------------------------------------------------------------- launch ---
extern "C" void kernel_launch(void* const* d_in, const int* in_sizes, int n_in,
                              void* d_out, int out_size, void* d_ws, size_t ws_size,
                              hipStream_t stream) {
  (void)in_sizes; (void)n_in; (void)out_size; (void)ws_size;
  const float* x      = (const float*)d_in[0];
  const float* qkv_w  = (const float*)d_in[1];
  const float* qkv_b  = (const float*)d_in[2];
  const float* lkA    = (const float*)d_in[3];
  const float* lkB    = (const float*)d_in[4];
  const float* lvA    = (const float*)d_in[5];
  const float* lvB    = (const float*)d_in[6];
  const float* proj_w = (const float*)d_in[7];
  const float* proj_b = (const float*)d_in[8];
  float* out = (float*)d_out;

  char* ws = (char*)d_ws;
  size_t off = 0;
  auto alloc = [&](size_t bytes) -> void* {
    void* p = ws + off;
    off += (bytes + 255) & ~(size_t)255;
    return p;
  };
  bf16_t* xb    = (bf16_t*)alloc((size_t)MN * Cc * 2);
  bf16_t* wqkv  = (bf16_t*)alloc((size_t)3 * Cc * Cc * 2);
  bf16_t* wkA   = (bf16_t*)alloc((size_t)RANK * Cc * 2);
  bf16_t* wkB   = (bf16_t*)alloc((size_t)Cc * RANK * 2);
  bf16_t* wvA   = (bf16_t*)alloc((size_t)RANK * Cc * 2);
  bf16_t* wvB   = (bf16_t*)alloc((size_t)Cc * RANK * 2);
  bf16_t* wproj = (bf16_t*)alloc((size_t)Cc * Cc * 2);
  float*  qkvf  = (float*) alloc((size_t)MN * 3 * Cc * 4);
  bf16_t* tk    = (bf16_t*)alloc((size_t)MN * RANK * 2);
  bf16_t* tv    = (bf16_t*)alloc((size_t)MN * RANK * 2);
  bf16_t* qh    = (bf16_t*)alloc((size_t)MN * Cc * 2);
  bf16_t* kh    = (bf16_t*)alloc((size_t)MN * Cc * 2);
  bf16_t* vh    = (bf16_t*)alloc((size_t)MN * Cc * 2);
  bf16_t* ao    = (bf16_t*)alloc((size_t)MN * Cc * 2);

  auto cvt = [&](const float* src, bf16_t* dst, int n) {
    cvt_f32_bf16<<<(n + 255) / 256, 256, 0, stream>>>(src, dst, n);
  };
  cvt(x,      xb,    MN * Cc);
  cvt(qkv_w,  wqkv,  3 * Cc * Cc);
  cvt(lkA,    wkA,   RANK * Cc);
  cvt(lkB,    wkB,   Cc * RANK);
  cvt(lvA,    wvA,   RANK * Cc);
  cvt(lvB,    wvB,   Cc * RANK);
  cvt(proj_w, wproj, Cc * Cc);

  dim3 blk(256);
  // QKV projection: qkv = x @ qkv_w^T + qkv_b   (fp32 out)
  gemm_bf16_wmma<<<dim3(MN / 128, (3 * Cc) / 64), blk, 0, stream>>>(
      xb, Cc, wqkv, Cc, qkvf, nullptr, 3 * Cc, qkv_b, 1.0f, Cc, 0);
  // LoRA K stage 1: tk = x @ lora_k_A^T   (bf16 out)
  gemm_bf16_wmma<<<dim3(MN / 128, RANK / 64), blk, 0, stream>>>(
      xb, Cc, wkA, Cc, nullptr, tk, RANK, nullptr, 1.0f, Cc, 0);
  // LoRA K stage 2: qkv[:, C:2C] += (tk @ lora_k_B^T) * (alpha/rank)
  gemm_bf16_wmma<<<dim3(MN / 128, Cc / 64), blk, 0, stream>>>(
      tk, RANK, wkB, RANK, qkvf + Cc, nullptr, 3 * Cc, nullptr, LORA_SCALE, RANK, 1);
  // LoRA V stage 1 + 2
  gemm_bf16_wmma<<<dim3(MN / 128, RANK / 64), blk, 0, stream>>>(
      xb, Cc, wvA, Cc, nullptr, tv, RANK, nullptr, 1.0f, Cc, 0);
  gemm_bf16_wmma<<<dim3(MN / 128, Cc / 64), blk, 0, stream>>>(
      tv, RANK, wvB, RANK, qkvf + 2 * Cc, nullptr, 3 * Cc, nullptr, LORA_SCALE, RANK, 1);

  // head split (+ fold softmax scale into Q)
  pack_qkv<<<(MN * Cc + 255) / 256, 256, 0, stream>>>(qkvf, qh, kh, vh);

  // attention
  flash_attn<<<dim3(Nn / 64, Bb * Hh), 128, 0, stream>>>(qh, kh, vh, ao);

  // output projection: out = ao @ proj_w^T + proj_b   (fp32 out)
  gemm_bf16_wmma<<<dim3(MN / 128, Cc / 64), blk, 0, stream>>>(
      ao, Cc, wproj, Cc, out, nullptr, Cc, proj_b, 1.0f, Cc, 0);
}
